// StaticCrossAttention_77412490543628
// MI455X (gfx1250) — matrix operand order repaired and checked by reading, not compiled, not run on hardware
//
#include <hip/hip_runtime.h>
#include <cstdint>
#include <cstddef>

// ---------------------------------------------------------------------------
// MI455X (gfx1250) cross-attention block, bf16 WMMA path, fp32 accumulate.
// LDS-staged GEMM + attention using async global->LDS copies (ASYNCcnt).
// ---------------------------------------------------------------------------

typedef __attribute__((ext_vector_type(16))) __bf16 v16bf;
typedef __attribute__((ext_vector_type(8)))  __bf16 v8bf;
typedef __attribute__((ext_vector_type(8)))  float  v8f;
typedef __attribute__((ext_vector_type(4)))  int    v4i;

union BFrag { v16bf v; v8bf h[2]; };

constexpr int   Bn    = 2;
constexpr int   Sn    = 4096;
constexpr int   Ln    = 512;
constexpr int   DIMn  = 5120;
constexpr int   Hn    = 40;
constexpr int   Dn    = 128;            // head dim
constexpr float EPSc  = 1e-6f;
constexpr float SCALEc = 0.08838834764831845f;  // 128^-0.5

#if __has_builtin(__builtin_amdgcn_global_load_async_to_lds_b128) && \
    __has_builtin(__builtin_amdgcn_s_wait_asynccnt)
#define USE_ASYNC_LDS 1
#else
#define USE_ASYNC_LDS 0
#endif

// 16-byte global -> LDS copy (async DMA when available, sync fallback)
__device__ __forceinline__ void copy16_g2l(const __bf16* g, __bf16* l) {
#if USE_ASYNC_LDS
  __builtin_amdgcn_global_load_async_to_lds_b128(
      (v4i*)(void*)g, (v4i*)(void*)l, 0, 0);
#else
  *(v8bf*)l = *(const v8bf*)g;
#endif
}
__device__ __forceinline__ void async_join() {
#if USE_ASYNC_LDS
  __builtin_amdgcn_s_wait_asynccnt(0);
#endif
}

__device__ __forceinline__ __bf16 f2bf(float f) {
  uint32_t u = __builtin_bit_cast(uint32_t, f);
  uint32_t r = u + 0x7FFFu + ((u >> 16) & 1u);   // round-to-nearest-even
  unsigned short hv = (unsigned short)(r >> 16);
  return __builtin_bit_cast(__bf16, hv);
}

// ---------------------------------------------------------------------------
// elementwise fp32 -> bf16
// ---------------------------------------------------------------------------
__global__ __launch_bounds__(256) void cast_f32_bf16_kernel(
    const float* __restrict__ x, __bf16* __restrict__ y, long long n) {
  long long i = (long long)blockIdx.x * blockDim.x + threadIdx.x;
  long long stride = (long long)gridDim.x * blockDim.x;
  for (; i < n; i += stride) y[i] = f2bf(x[i]);
}

// ---------------------------------------------------------------------------
// W[K,N] fp32 -> Wt[N,K] bf16  (so GEMM B-fragments are contiguous in K)
// ---------------------------------------------------------------------------
__global__ __launch_bounds__(256) void transpose_cast_kernel(
    const float* __restrict__ W, __bf16* __restrict__ Wt, int K, int N) {
  __shared__ float tile[32][33];
  int n0 = blockIdx.x * 32, k0 = blockIdx.y * 32;
  int tx = threadIdx.x, ty = threadIdx.y;
  for (int i = ty; i < 32; i += 8)
    tile[i][tx] = W[(size_t)(k0 + i) * N + (n0 + tx)];
  __syncthreads();
  for (int i = ty; i < 32; i += 8)
    Wt[(size_t)(n0 + i) * K + (k0 + tx)] = f2bf(tile[tx][i]);
}

// ---------------------------------------------------------------------------
// C[M,N] = A[M,K](bf16) * Bt[N,K]^T(bf16) + bias, fp32 out.
// Block = 8 waves = 128x128 tile, K staged 64-wide through double-buffered
// LDS via async global->LDS DMA; wave tile = 64x32 (4x2 WMMA tiles).
// ---------------------------------------------------------------------------
__global__ __launch_bounds__(256) void gemm_bf16_wmma_kernel(
    const __bf16* __restrict__ A, const __bf16* __restrict__ Bt,
    const float* __restrict__ bias, float* __restrict__ C,
    int M, int N, int K) {
  constexpr int KT  = 64;    // K per stage
  constexpr int LDP = 72;    // padded LDS row stride (bf16 elems), 16B multiple
  __shared__ alignas(16) __bf16 Atile[2][128][LDP];
  __shared__ alignas(16) __bf16 Btile[2][128][LDP];

  const int tid  = threadIdx.x;
  const int lane = tid & 31;
  const int wave = tid >> 5;
  const int l16  = lane & 15;
  const int hf   = lane >> 4;
  const int nBlk = N >> 7;
  const int bm = blockIdx.x / nBlk;
  const int bn = blockIdx.x % nBlk;
  const int rowA0 = bm * 128;
  const int rowB0 = bn * 128;
  const int wm = (wave >> 2) * 64;     // wave M offset within tile
  const int wn = (wave & 3) * 32;     // wave N offset within tile
  (void)M;

  v8f acc[4][2] = {};

  // Stage copy: 128x64 bf16 per tile = 1024 16B-chunks; 4 per thread per tile.
  auto issue_stage = [&](int buf, int kc) {
#pragma unroll
    for (int j = 0; j < 4; ++j) {
      int c = tid + 256 * j;
      int r = c >> 3, col = (c & 7) * 8;
      copy16_g2l(A  + (size_t)(rowA0 + r) * K + kc + col, &Atile[buf][r][col]);
      copy16_g2l(Bt + (size_t)(rowB0 + r) * K + kc + col, &Btile[buf][r][col]);
    }
  };

  issue_stage(0, 0);
  for (int kc = 0; kc < K; kc += KT) {
    const int buf = (kc / KT) & 1;
    async_join();
    __syncthreads();                       // stage `buf` visible to all waves
    if (kc + KT < K) issue_stage(buf ^ 1, kc + KT);   // DMA overlaps compute
#pragma unroll
    for (int kk = 0; kk < KT; kk += 32) {
      BFrag a[4], b[2];
#pragma unroll
      for (int mt = 0; mt < 4; ++mt) {
        const __bf16* p = &Atile[buf][wm + mt * 16 + l16][kk];
        a[mt].h[0] = *(const v8bf*)(p + hf * 8);
        a[mt].h[1] = *(const v8bf*)(p + 16 + hf * 8);
      }
#pragma unroll
      for (int nt = 0; nt < 2; ++nt) {
        const __bf16* p = &Btile[buf][wn + nt * 16 + l16][kk + hf * 8];
        b[nt].h[0] = *(const v8bf*)(p);
        b[nt].h[1] = *(const v8bf*)(p + 16);
      }
#pragma unroll
      for (int mt = 0; mt < 4; ++mt)
#pragma unroll
        for (int nt = 0; nt < 2; ++nt)
          acc[mt][nt] = __builtin_amdgcn_wmma_f32_16x16x32_bf16(
              false, a[mt].v, false, b[nt].v, (short)0, acc[mt][nt], false, false);
    }
    __syncthreads();                       // all reads done before next DMA lands
  }

#pragma unroll
  for (int mt = 0; mt < 4; ++mt)
#pragma unroll
    for (int nt = 0; nt < 2; ++nt) {
      int col = bn * 128 + wn + nt * 16 + l16;
      float bv = bias[col];
#pragma unroll
      for (int r = 0; r < 8; ++r) {
        int row = bm * 128 + wm + mt * 16 + r + hf * 8;
        C[(size_t)row * N + col] = acc[mt][nt][r] + bv;
      }
    }
}

// ---------------------------------------------------------------------------
// Per-row RMSNorm of biased Q projection -> bf16
// ---------------------------------------------------------------------------
__global__ __launch_bounds__(256) void rmsnorm_q_kernel(
    const float* __restrict__ x, const float* __restrict__ g,
    __bf16* __restrict__ y) {
  __shared__ float red[8];
  const size_t row = blockIdx.x;
  const float* xr = x + row * DIMn;
  float s = 0.f;
  for (int i = threadIdx.x; i < DIMn; i += 256) { float v = xr[i]; s += v * v; }
  for (int off = 16; off; off >>= 1) s += __shfl_xor(s, off, 32);
  if ((threadIdx.x & 31) == 0) red[threadIdx.x >> 5] = s;
  __syncthreads();
  float tot = 0.f;
#pragma unroll
  for (int i = 0; i < 8; ++i) tot += red[i];
  float inv = rsqrtf(tot / (float)DIMn + EPSc);
  __bf16* yr = y + row * DIMn;
  for (int i = threadIdx.x; i < DIMn; i += 256) yr[i] = f2bf(xr[i] * inv * g[i]);
}

// ---------------------------------------------------------------------------
// KV post: K = rmsnorm(kv[:,:DIM])*gk -> bf16 [B,H,L,D];  V -> bf16 [B,H,D,L]
// ---------------------------------------------------------------------------
__global__ __launch_bounds__(256) void kv_post_kernel(
    const float* __restrict__ kv, const float* __restrict__ gk,
    __bf16* __restrict__ kout, __bf16* __restrict__ vt) {
  __shared__ float red[8];
  const int row = blockIdx.x;                    // b*L + lpos
  const int b = row / Ln, lpos = row % Ln;
  const float* kr = kv + (size_t)row * (2 * DIMn);
  float s = 0.f;
  for (int i = threadIdx.x; i < DIMn; i += 256) { float v = kr[i]; s += v * v; }
  for (int off = 16; off; off >>= 1) s += __shfl_xor(s, off, 32);
  if ((threadIdx.x & 31) == 0) red[threadIdx.x >> 5] = s;
  __syncthreads();
  float tot = 0.f;
#pragma unroll
  for (int i = 0; i < 8; ++i) tot += red[i];
  float inv = rsqrtf(tot / (float)DIMn + EPSc);
  for (int i = threadIdx.x; i < DIMn; i += 256) {
    int hh = i >> 7, d = i & 127;
    size_t bh = (size_t)b * Hn + hh;
    kout[(bh * Ln + lpos) * Dn + d] = f2bf(kr[i] * inv * gk[i]);
    vt[(bh * Dn + d) * Ln + lpos]   = f2bf(kr[DIMn + i]);
  }
}

// ---------------------------------------------------------------------------
// Flash-style attention core. grid = (S/128, B*H), block = 256 (8 waves).
// Each wave owns 16 query rows. K/V chunks (32 positions) staged in LDS once
// per block via async DMA and shared by all 8 waves; online softmax.
// ---------------------------------------------------------------------------
__global__ __launch_bounds__(256) void attention_kernel(
    const __bf16* __restrict__ q,   // [B*S, DIM]
    const __bf16* __restrict__ k,   // [B,H,L,D]
    const __bf16* __restrict__ vt,  // [B,H,D,L]
    __bf16* __restrict__ o) {       // [B*S, DIM]
  __shared__ alignas(16) __bf16 Ktile[32][136];    // [pos][d], padded
  __shared__ alignas(16) __bf16 Vtile[128][40];    // [d][pos], padded
  __shared__ alignas(16) __bf16 plds[8][16][40];   // per-wave P re-layout
  const int tid  = threadIdx.x;
  const int lane = tid & 31;
  const int wave = tid >> 5;
  const int l16 = lane & 15, hf = lane >> 4;
  const int bh = blockIdx.y;
  const int b = bh / Hn, h = bh % Hn;
  const int s0 = blockIdx.x * 128 + wave * 16;

  // Q fragments: 4 K-chunks of 32 over head dim (A-matrix layout)
  BFrag qf[4];
  const __bf16* qrow = q + ((size_t)b * Sn + s0 + l16) * DIMn + h * Dn;
#pragma unroll
  for (int c = 0; c < 4; ++c) {
    qf[c].h[0] = *(const v8bf*)(qrow + c * 32 + hf * 8);
    qf[c].h[1] = *(const v8bf*)(qrow + c * 32 + 16 + hf * 8);
  }

  v8f oacc[8] = {};
  float mrow[8], lsum[8];
#pragma unroll
  for (int r = 0; r < 8; ++r) { mrow[r] = -3.0e38f; lsum[r] = 0.f; }

  const __bf16* kbase = k + (size_t)bh * Ln * Dn;
  const __bf16* vbase = vt + (size_t)bh * Dn * Ln;

  for (int p0 = 0; p0 < Ln; p0 += 32) {
    // ---- stage K[32 x 128] and Vt[128 x 32] chunks in LDS (shared) ----
#pragma unroll
    for (int j = 0; j < 2; ++j) {
      int c = tid + 256 * j;
      int kr = c >> 4, kcol = (c & 15) * 8;        // K: 32 rows x 128 cols
      copy16_g2l(kbase + (size_t)(p0 + kr) * Dn + kcol, &Ktile[kr][kcol]);
      int vr = c >> 2, vcol = (c & 3) * 8;         // V: 128 rows x 32 cols
      copy16_g2l(vbase + (size_t)vr * Ln + p0 + vcol, &Vtile[vr][vcol]);
    }
    async_join();
    __syncthreads();

    // ---- scores for two 16-wide position chunks (Q @ K^T) ----
    v8f sc0 = {}, sc1 = {};
#pragma unroll
    for (int c = 0; c < 4; ++c) {
      BFrag b0, b1;
      const __bf16* p0k = &Ktile[l16][c * 32 + hf * 8];
      const __bf16* p1k = &Ktile[16 + l16][c * 32 + hf * 8];
      b0.h[0] = *(const v8bf*)(p0k);
      b0.h[1] = *(const v8bf*)(p0k + 16);
      b1.h[0] = *(const v8bf*)(p1k);
      b1.h[1] = *(const v8bf*)(p1k + 16);
      sc0 = __builtin_amdgcn_wmma_f32_16x16x32_bf16(false, qf[c].v, false, b0.v,
                                                    (short)0, sc0, false, false);
      sc1 = __builtin_amdgcn_wmma_f32_16x16x32_bf16(false, qf[c].v, false, b1.v,
                                                    (short)0, sc1, false, false);
    }
    // ---- online softmax; C-layout row = r + 8*hf, reduce over 16 lanes ----
#pragma unroll
    for (int r = 0; r < 8; ++r) {
      float x0 = sc0[r] * SCALEc;
      float x1 = sc1[r] * SCALEc;
      float cm = fmaxf(x0, x1);
#pragma unroll
      for (int off = 1; off < 16; off <<= 1) cm = fmaxf(cm, __shfl_xor(cm, off, 32));
      float nm   = fmaxf(mrow[r], cm);
      float corr = __expf(mrow[r] - nm);
      float p1 = __expf(x0 - nm);
      float p2 = __expf(x1 - nm);
      float rs = p1 + p2;
#pragma unroll
      for (int off = 1; off < 16; off <<= 1) rs += __shfl_xor(rs, off, 32);
      lsum[r] = lsum[r] * corr + rs;
      mrow[r] = nm;
#pragma unroll
      for (int t = 0; t < 8; ++t) oacc[t][r] *= corr;
      plds[wave][r + 8 * hf][l16]      = f2bf(p1);
      plds[wave][r + 8 * hf][16 + l16] = f2bf(p2);
    }
    asm volatile("" ::: "memory");   // keep LDS stores before re-layout loads
    // ---- P re-layout (C -> A fragment) through LDS ----
    BFrag pf;
    {
      const __bf16* pr = &plds[wave][l16][0];
      pf.h[0] = *(const v8bf*)(pr + hf * 8);
      pf.h[1] = *(const v8bf*)(pr + 16 + hf * 8);
    }
    // ---- P @ V: 8 d-tiles of 16 ----
#pragma unroll
    for (int t = 0; t < 8; ++t) {
      const __bf16* vr = &Vtile[t * 16 + l16][hf * 8];
      BFrag vb;
      vb.h[0] = *(const v8bf*)(vr);
      vb.h[1] = *(const v8bf*)(vr + 16);
      oacc[t] = __builtin_amdgcn_wmma_f32_16x16x32_bf16(
          false, pf.v, false, vb.v, (short)0, oacc[t], false, false);
    }
    __syncthreads();                 // all waves done with K/V tiles
  }

#pragma unroll
  for (int t = 0; t < 8; ++t) {
    int col = h * Dn + t * 16 + l16;
#pragma unroll
    for (int r = 0; r < 8; ++r) {
      int srow = s0 + r + 8 * hf;
      o[((size_t)b * Sn + srow) * DIMn + col] = f2bf(oacc[t][r] / lsum[r]);
    }
  }
}

// ---------------------------------------------------------------------------
// Host-side launch
// ---------------------------------------------------------------------------
extern "C" void kernel_launch(void* const* d_in, const int* in_sizes, int n_in,
                              void* d_out, int out_size, void* d_ws, size_t ws_size,
                              hipStream_t stream) {
  (void)in_sizes; (void)n_in; (void)out_size; (void)ws_size;
  const float* hs  = (const float*)d_in[0];
  const float* ctx = (const float*)d_in[1];
  const float* Wq  = (const float*)d_in[2];
  const float* bq  = (const float*)d_in[3];
  const float* Wkv = (const float*)d_in[4];
  const float* bkv = (const float*)d_in[5];
  const float* gq  = (const float*)d_in[6];
  const float* gk  = (const float*)d_in[7];
  const float* Wo  = (const float*)d_in[8];
  const float* bo  = (const float*)d_in[9];
  float* out = (float*)d_out;

  char* ws = (char*)d_ws;
  size_t off = 0;
  auto alloc = [&](size_t bytes) -> void* {
    void* p = ws + off;
    off = (off + bytes + 255) & ~(size_t)255;
    return p;
  };

  const size_t nQ  = (size_t)Bn * Sn * DIMn;       // 8192*5120
  const size_t nC  = (size_t)Bn * Ln * DIMn;       // 1024*5120
  const size_t nKV = (size_t)Bn * Ln * 2 * DIMn;   // 1024*10240
  const size_t nHD = (size_t)Bn * Hn * Ln * Dn;    // 80*512*128

  __bf16* hs_bf   = (__bf16*)alloc(nQ * 2);
  __bf16* ctx_bf  = (__bf16*)alloc(nC * 2);
  __bf16* WqT     = (__bf16*)alloc((size_t)DIMn * DIMn * 2);
  __bf16* WkvT    = (__bf16*)alloc((size_t)DIMn * 2 * DIMn * 2);
  __bf16* WoT     = (__bf16*)alloc((size_t)DIMn * DIMn * 2);
  float*  qtmp    = (float*)alloc(nQ * 4);
  float*  kvtmp   = (float*)alloc(nKV * 4);
  __bf16* q_bf    = (__bf16*)alloc(nQ * 2);
  __bf16* k_bf    = (__bf16*)alloc(nHD * 2);
  __bf16* vt_bf   = (__bf16*)alloc(nHD * 2);
  __bf16* attn_bf = (__bf16*)qtmp;   // overlay: qtmp dead after rmsnorm_q

  // 1) casts
  cast_f32_bf16_kernel<<<4096, 256, 0, stream>>>(hs, hs_bf, (long long)nQ);
  cast_f32_bf16_kernel<<<1024, 256, 0, stream>>>(ctx, ctx_bf, (long long)nC);
  transpose_cast_kernel<<<dim3(DIMn / 32, DIMn / 32), dim3(32, 8), 0, stream>>>(
      Wq, WqT, DIMn, DIMn);
  transpose_cast_kernel<<<dim3(2 * DIMn / 32, DIMn / 32), dim3(32, 8), 0, stream>>>(
      Wkv, WkvT, DIMn, 2 * DIMn);
  transpose_cast_kernel<<<dim3(DIMn / 32, DIMn / 32), dim3(32, 8), 0, stream>>>(
      Wo, WoT, DIMn, DIMn);

  // 2) Q = hidden @ Wq + bq ; then RMSNorm*gq -> bf16
  gemm_bf16_wmma_kernel<<<(Bn * Sn / 128) * (DIMn / 128), 256, 0, stream>>>(
      hs_bf, WqT, bq, qtmp, Bn * Sn, DIMn, DIMn);
  rmsnorm_q_kernel<<<Bn * Sn, 256, 0, stream>>>(qtmp, gq, q_bf);

  // 3) KV = context @ Wkv + bkv ; split + norm K, transpose V
  gemm_bf16_wmma_kernel<<<(Bn * Ln / 128) * (2 * DIMn / 128), 256, 0, stream>>>(
      ctx_bf, WkvT, bkv, kvtmp, Bn * Ln, 2 * DIMn, DIMn);
  kv_post_kernel<<<Bn * Ln, 256, 0, stream>>>(kvtmp, gk, k_bf, vt_bf);

  // 4) attention core
  attention_kernel<<<dim3(Sn / 128, Bn * Hn), 256, 0, stream>>>(
      q_bf, k_bf, vt_bf, attn_bf);

  // 5) out = attn @ Wo + bo (fp32)
  gemm_bf16_wmma_kernel<<<(Bn * Sn / 128) * (DIMn / 128), 256, 0, stream>>>(
      attn_bf, WoT, bo, out, Bn * Sn, DIMn, DIMn);
}
